// TimeMixing_21938692948062
// MI455X (gfx1250) — matrix operand order repaired and checked by reading, not compiled, not run on hardware
//
#include <hip/hip_runtime.h>

typedef __attribute__((ext_vector_type(16))) _Float16 v16h;
typedef __attribute__((ext_vector_type(8)))  _Float16 v8h;
typedef __attribute__((ext_vector_type(8)))  float    v8f;
typedef __attribute__((ext_vector_type(4)))  unsigned int v4u;
typedef __attribute__((ext_vector_type(8)))  int v8i_;
typedef __attribute__((ext_vector_type(4)))  int v4i_;

#define BB    2
#define TT    2048
#define DD    1024
#define HH    16
#define HDIM  64
#define RLORA 64
#define MROWS (BB*TT)   /* 4096 */
#define BHQ   (BB*HH)   /* 32 (b,h) pairs */
#define CS    64        /* scan chunk size (timesteps) */
#define NC    (TT/CS)   /* 32 chunks */
#define KSTEP 64
#define LDST  72        /* LDS row stride in halfs: 64 data + 8 pad = 144B (16B aligned) */

// ---------------- CDNA5 async / tensor copy helpers ----------------

__device__ __forceinline__ void g2l_async_b128(const void* gaddr, uint32_t lds_byte_addr) {
  asm volatile("global_load_async_to_lds_b128 %0, %1, off"
               :: "v"(lds_byte_addr), "v"(gaddr)
               : "memory");
}
__device__ __forceinline__ void wait_async0() {
  asm volatile("s_wait_asynccnt 0" ::: "memory");
}
__device__ __forceinline__ uint32_t lds_addr32(const void* p) {
  // generic pointers to LDS carry the LDS byte address in the low 32 bits (ISA §10.2)
  return (uint32_t)(uintptr_t)p;
}

// One-shot 2D Tensor-DMA: tile1 rows x tile0 elems (2B each), global row stride `stride0` elems,
// into LDS at lds_addr with hardware padding of 16B after every 128B row (matches LDST=72 halfs).
__device__ __forceinline__ void tdm_load_2d_b16(const void* gbase, uint32_t lds_addr,
                                                uint32_t dim0, uint32_t dim1,
                                                uint32_t stride0, uint32_t tile0, uint32_t tile1) {
  const uint64_t ga = (uint64_t)gbase;
  v4u g0;
  g0[0] = 1u;                                         // count=1 (valid), user mode
  g0[1] = lds_addr;                                   // lds_addr[31:0]
  g0[2] = (uint32_t)(ga & 0xffffffffu);               // global_addr[31:0]
  g0[3] = (uint32_t)((ga >> 32) & 0x1ffffffu)         // global_addr[56:32]
          | (2u << 30);                               // type=2 ("image")
  v8i_ g1;
  g1[0] = (int)((1u << 16)                            // data_size: 1 -> 2 bytes
                | (1u << 20)                          // pad_enable
                | (4u << 22)                          // pad_interval: 32 DWORDs (128B)
                | (3u << 25));                        // pad_amount: 4 DWORDs (16B)
  g1[1] = (int)((dim0 & 0xffffu) << 16);              // atomic_barrier_addr=0 | tensor_dim0 lo16
  g1[2] = (int)((dim0 >> 16) | ((dim1 & 0xffffu) << 16));   // dim0 hi16 | tensor_dim1 lo16
  g1[3] = (int)((dim1 >> 16) | (tile0 << 16));        // dim1 hi16 | tile_dim0
  g1[4] = (int)(tile1 & 0xffffu);                     // tile_dim1 | tile_dim2=0
  g1[5] = (int)stride0;                               // tensor_dim0_stride[31:0]
  g1[6] = 0;                                          // stride0 hi16 | dim1_stride lo16
  g1[7] = 0;
  v4i_ g2; g2[0] = 0; g2[1] = 0; g2[2] = 0; g2[3] = 0;
  v4i_ g3; g3[0] = 0; g3[1] = 0; g3[2] = 0; g3[3] = 0;
  v8i_ g4;
#pragma unroll
  for (int e = 0; e < 8; ++e) g4[e] = 0;
  __builtin_amdgcn_tensor_load_to_lds(g0, g1, g2, g3, g4, 0);
}

// ---------------- elementwise kernels ----------------

__global__ void k_shift_mix(const float* __restrict__ x,
                            const float* __restrict__ mu_w, const float* __restrict__ mu_r,
                            const float* __restrict__ mu_k, const float* __restrict__ mu_v,
                            const float* __restrict__ mu_g,
                            _Float16* __restrict__ xw, _Float16* __restrict__ xr,
                            _Float16* __restrict__ xk, _Float16* __restrict__ xv,
                            _Float16* __restrict__ xg) {
  int i = blockIdx.x * blockDim.x + threadIdx.x;   // < MROWS*DD
  int d  = i & (DD - 1);
  int bt = i >> 10;
  int t  = bt & (TT - 1);
  float xi = x[i];
  float xp = (t > 0) ? x[i - DD] : 0.f;
  float xx = xp - xi;
  xw[i] = (_Float16)(xi + xx * mu_w[d]);
  xr[i] = (_Float16)(xi + xx * mu_r[d]);
  xk[i] = (_Float16)(xi + xx * mu_k[d]);
  xv[i] = (_Float16)(xi + xx * mu_v[d]);
  xg[i] = (_Float16)(xi + xx * mu_g[d]);
}

// f32 (R x C, row-major) -> f16 transposed (C x R): GEMM B-operands become [N][K]
__global__ void k_f32_to_f16_T(const float* __restrict__ src, _Float16* __restrict__ dst,
                               int R, int Ccols) {
  int i = blockIdx.x * blockDim.x + threadIdx.x;
  if (i < R * Ccols) {
    int r = i / Ccols;
    int c = i - r * Ccols;
    dst[(size_t)c * R + r] = (_Float16)src[i];
  }
}

__global__ void k_tanh_h(const float* __restrict__ hraw, _Float16* __restrict__ h16) {
  int i = blockIdx.x * blockDim.x + threadIdx.x;
  if (i < MROWS * RLORA) h16[i] = (_Float16)tanhf(hraw[i]);
}

// ---------------- WMMA GEMM: C[M,N] = A[M,K](f16,row) * BT[N,K](f16,row = B^T) --------------
// 256 threads (8 waves), tile 128(M) x 16*NTILES(N), K-step 64, double-buffered LDS.
// A staged via per-lane GLOBAL_LOAD_ASYNC_TO_LDS_B128 (ASYNCcnt); B staged via one
// TENSOR_LOAD_TO_LDS issued by wave 0 (TENSORcnt, HW row padding).

__device__ inline v8f vzero8() {
  v8f z;
#pragma unroll
  for (int e = 0; e < 8; ++e) z[e] = 0.f;
  return z;
}

__device__ __forceinline__ void tile_async_copy(const _Float16* __restrict__ src, int r0, int ld,
                                                int k0, _Float16* lds_base, int tid,
                                                int valid_rows) {
#pragma unroll
  for (int i = 0; i < 4; ++i) {
    int idx = tid + i * 256;          // 0..1023 chunks of 8 halfs
    int row = idx >> 3;
    int seg = idx & 7;
    if (row < valid_rows) {
      const void* g = src + (size_t)(r0 + row) * ld + (size_t)k0 + seg * 8;
      g2l_async_b128(g, lds_addr32(lds_base + row * LDST + seg * 8));
    }
  }
}

template <int NTILES>
__global__ __launch_bounds__(256) void k_gemm_wmma(const _Float16* __restrict__ A,
                                                   const _Float16* __restrict__ BT,
                                                   float* __restrict__ C,
                                                   int M, int N, int K) {
  __shared__ __align__(16) _Float16 As[2][128 * LDST];
  __shared__ __align__(16) _Float16 Bs[2][128 * LDST];

  const int tid  = threadIdx.x;
  const int lane = tid & 31;
  const int wv   = tid >> 5;
  const int m0   = blockIdx.y * 128;
  const int n0   = blockIdx.x * 128;
  const int bn   = (N - n0 < 128) ? (N - n0) : 128;

  v8f acc[NTILES];
#pragma unroll
  for (int i = 0; i < NTILES; ++i) acc[i] = vzero8();

  // prologue: stage first K-tile (A: async-to-LDS; B: tensor DMA from wave 0)
  tile_async_copy(A, m0, K, 0, As[0], tid, 128);
  if (wv == 0) {
    tdm_load_2d_b16(BT + (size_t)n0 * K, lds_addr32(&Bs[0][0]),
                    (uint32_t)K, (uint32_t)bn, (uint32_t)K, KSTEP, (uint32_t)bn);
  }
  wait_async0();
  if (wv == 0) __builtin_amdgcn_s_wait_tensorcnt(0);
  __syncthreads();

  int p = 0;
  for (int k0 = 0; k0 < K; k0 += KSTEP) {
    if (k0 + KSTEP < K) {
      tile_async_copy(A, m0, K, k0 + KSTEP, As[p ^ 1], tid, 128);
      if (wv == 0) {
        tdm_load_2d_b16(BT + (size_t)n0 * K + (k0 + KSTEP), lds_addr32(&Bs[p ^ 1][0]),
                        (uint32_t)K, (uint32_t)bn, (uint32_t)K, KSTEP, (uint32_t)bn);
      }
    }

#pragma unroll
    for (int kh = 0; kh < 2; ++kh) {    // two 16x16x32 sub-steps per 64-wide K tile
      // A fragment: lane<16 -> M=lane, K{0..7,16..23}; lane>=16 -> K{8..15,24..31} (+kh*32)
      const int arow = wv * 16 + (lane & 15);
      const int akb  = kh * 32 + ((lane < 16) ? 0 : 8);
      v16h afrag;
      {
        v8h lo = *(const v8h*)(&As[p][arow * LDST + akb]);
        v8h hi = *(const v8h*)(&As[p][arow * LDST + akb + 16]);
#pragma unroll
        for (int e = 0; e < 8; ++e) { afrag[e] = lo[e]; afrag[e + 8] = hi[e]; }
      }
      // batch-load ALL B fragments, then fire WMMAs back-to-back (one dscnt wait, XDL-friendly)
      v16h bfr[NTILES];
#pragma unroll
      for (int nt = 0; nt < NTILES; ++nt) {
        const int brow = nt * 16 + (lane & 15);
        const int bkb  = kh * 32 + ((lane < 16) ? 0 : 8);
        v8h blo = *(const v8h*)(&Bs[p][brow * LDST + bkb]);
        v8h bhi = *(const v8h*)(&Bs[p][brow * LDST + bkb + 16]);
#pragma unroll
        for (int e = 0; e < 8; ++e) { bfr[nt][e] = blo[e]; bfr[nt][e + 8] = bhi[e]; }
      }
#pragma unroll
      for (int nt = 0; nt < NTILES; ++nt) {
        acc[nt] = __builtin_amdgcn_wmma_f32_16x16x32_f16(
            false, afrag, false, bfr[nt], (short)0, acc[nt], false, false);
      }
    }

    wait_async0();
    if (wv == 0) __builtin_amdgcn_s_wait_tensorcnt(0);
    __syncthreads();
    p ^= 1;
  }

  const int mrow  = m0 + wv * 16 + ((lane < 16) ? 0 : 8);
  const int ncol0 = n0 + (lane & 15);
#pragma unroll
  for (int nt = 0; nt < NTILES; ++nt) {
#pragma unroll
    for (int r = 0; r < 8; ++r) {
      C[(size_t)(mrow + r) * N + (ncol0 + nt * 16)] = acc[nt][r];
    }
  }
}

// ---------------- chunked WKV scan (3 passes) ----------------
// Per (i,j): S_t = w_t[i]*S_{t-1} + k_t[i]*v_t[j] (linear, diagonal decay).
// Pass A: 64-step chunk partials + per-row decay products (32x32 blocks).
// Pass B: 32-step prefix combine (32 blocks).  Pass C: replay chunks + GN + SiLU-gate.

__global__ __launch_bounds__(64) void k_scan_partial(const float* __restrict__ k32,
                                                     const float* __restrict__ v32,
                                                     const float* __restrict__ zlora,
                                                     const float* __restrict__ w0,
                                                     float* __restrict__ Spart,
                                                     float* __restrict__ Pprod) {
  const int bh = blockIdx.x, c = blockIdx.y;
  const int b = bh >> 4, h = bh & 15;
  const int j = threadIdx.x;
  __shared__ __align__(16) float ks[HDIM], dcy[HDIM];
  const float w0j = w0[h * HDIM + j];

  float S[HDIM];
#pragma unroll
  for (int i = 0; i < HDIM; ++i) S[i] = 0.f;
  float pj = 1.f;

  const size_t base = (size_t)b * TT * DD + (size_t)h * HDIM;
  for (int tt = 0; tt < CS; ++tt) {
    const size_t off = base + (size_t)(c * CS + tt) * DD;
    const float dj = expf(-expf(w0j + zlora[off + j]));
    ks[j]  = k32[off + j];
    dcy[j] = dj;
    const float vj = v32[off + j];
    pj *= dj;
    __syncthreads();
#pragma unroll
    for (int q = 0; q < HDIM / 4; ++q) {
      const float4 k4 = *(const float4*)&ks[q * 4];
      const float4 d4 = *(const float4*)&dcy[q * 4];
      S[q*4+0] = fmaf(d4.x, S[q*4+0], k4.x * vj);
      S[q*4+1] = fmaf(d4.y, S[q*4+1], k4.y * vj);
      S[q*4+2] = fmaf(d4.z, S[q*4+2], k4.z * vj);
      S[q*4+3] = fmaf(d4.w, S[q*4+3], k4.w * vj);
    }
    __syncthreads();
  }
  float* sp = Spart + ((size_t)bh * NC + c) * HDIM * HDIM;
#pragma unroll
  for (int i = 0; i < HDIM; ++i) sp[i * HDIM + j] = S[i];
  Pprod[((size_t)bh * NC + c) * HDIM + j] = pj;
}

__global__ __launch_bounds__(64) void k_scan_prefix(const float* __restrict__ Spart,
                                                    const float* __restrict__ Pprod,
                                                    float* __restrict__ Sinit) {
  const int bh = blockIdx.x;
  const int j  = threadIdx.x;
  __shared__ __align__(16) float ps[HDIM];
  float Cst[HDIM];
#pragma unroll
  for (int i = 0; i < HDIM; ++i) Cst[i] = 0.f;
  for (int c = 0; c < NC; ++c) {
    const size_t tile = ((size_t)bh * NC + c) * HDIM * HDIM;
    ps[j] = Pprod[((size_t)bh * NC + c) * HDIM + j];
    __syncthreads();
    const float* sp = Spart + tile;
    float* si = Sinit + tile;
#pragma unroll
    for (int i = 0; i < HDIM; ++i) {
      si[i * HDIM + j] = Cst[i];                       // initial state seen by chunk c
      Cst[i] = fmaf(ps[i], Cst[i], sp[i * HDIM + j]);  // carry to next chunk
    }
    __syncthreads();
  }
}

__global__ __launch_bounds__(64) void k_scan_final(const float* __restrict__ r32,
                                                   const float* __restrict__ k32,
                                                   const float* __restrict__ v32,
                                                   const float* __restrict__ zlora,
                                                   const float* __restrict__ graw,
                                                   const float* __restrict__ w0,
                                                   const float* __restrict__ u,
                                                   const float* __restrict__ gamma,
                                                   const float* __restrict__ beta,
                                                   const float* __restrict__ Sinit,
                                                   _Float16* __restrict__ yg16) {
  const int bh = blockIdx.x, c = blockIdx.y;
  const int b = bh >> 4, h = bh & 15;
  const int j = threadIdx.x;
  __shared__ __align__(16) float rs[HDIM], ks[HDIM], dcy[HDIM], us[HDIM], yred[HDIM];
  us[j] = u[h * HDIM + j];
  const float w0j = w0[h * HDIM + j];
  const float gmj = gamma[h * HDIM + j];
  const float btj = beta[h * HDIM + j];

  float S[HDIM];
  const float* si = Sinit + ((size_t)bh * NC + c) * HDIM * HDIM;
#pragma unroll
  for (int i = 0; i < HDIM; ++i) S[i] = si[i * HDIM + j];
  __syncthreads();

  const size_t base = (size_t)b * TT * DD + (size_t)h * HDIM;
  for (int tt = 0; tt < CS; ++tt) {
    const size_t off = base + (size_t)(c * CS + tt) * DD;
    rs[j]  = r32[off + j];
    ks[j]  = k32[off + j];
    dcy[j] = expf(-expf(w0j + zlora[off + j]));
    const float vj = v32[off + j];
    if (tt + 4 < CS) {
      __builtin_prefetch(&r32[off + 4 * DD + j], 0, 0);
      __builtin_prefetch(&k32[off + 4 * DD + j], 0, 0);
      __builtin_prefetch(&v32[off + 4 * DD + j], 0, 0);
      __builtin_prefetch(&zlora[off + 4 * DD + j], 0, 0);
    }
    __syncthreads();

    float y0 = 0.f, y1 = 0.f, y2 = 0.f, y3 = 0.f;
    float a0 = 0.f, a1 = 0.f, a2 = 0.f, a3 = 0.f;
#pragma unroll
    for (int q = 0; q < HDIM / 4; ++q) {
      const float4 r4 = *(const float4*)&rs[q * 4];
      const float4 k4 = *(const float4*)&ks[q * 4];
      const float4 d4 = *(const float4*)&dcy[q * 4];
      const float4 u4 = *(const float4*)&us[q * 4];
      y0 = fmaf(r4.x, S[q*4+0], y0); a0 = fmaf(r4.x*u4.x, k4.x, a0); S[q*4+0] = fmaf(d4.x, S[q*4+0], k4.x*vj);
      y1 = fmaf(r4.y, S[q*4+1], y1); a1 = fmaf(r4.y*u4.y, k4.y, a1); S[q*4+1] = fmaf(d4.y, S[q*4+1], k4.y*vj);
      y2 = fmaf(r4.z, S[q*4+2], y2); a2 = fmaf(r4.z*u4.z, k4.z, a2); S[q*4+2] = fmaf(d4.z, S[q*4+2], k4.z*vj);
      y3 = fmaf(r4.w, S[q*4+3], y3); a3 = fmaf(r4.w*u4.w, k4.w, a3); S[q*4+3] = fmaf(d4.w, S[q*4+3], k4.w*vj);
    }
    const float y = ((y0 + y1) + (y2 + y3)) + (((a0 + a1) + (a2 + a3)) * vj);

    yred[j] = y;
    __syncthreads();
    float m0 = 0.f, m1 = 0.f, s0 = 0.f, s1 = 0.f;
#pragma unroll
    for (int q = 0; q < HDIM / 4; ++q) {
      const float4 y4 = *(const float4*)&yred[q * 4];
      m0 += y4.x + y4.y; m1 += y4.z + y4.w;
      s0 = fmaf(y4.x, y4.x, s0); s0 = fmaf(y4.y, y4.y, s0);
      s1 = fmaf(y4.z, y4.z, s1); s1 = fmaf(y4.w, y4.w, s1);
    }
    const float m   = (m0 + m1) * (1.f / HDIM);
    const float var = (s0 + s1) * (1.f / HDIM) - m * m;
    const float yn  = (y - m) * rsqrtf(var + 1e-5f);
    const float gr  = graw[off + j];
    const float gv  = gr / (1.f + expf(-gr));     // fused SiLU
    yg16[off + j] = (_Float16)((yn * gmj + btj) * gv);
    __syncthreads();
  }
}

// ---------------- launcher ----------------

extern "C" void kernel_launch(void* const* d_in, const int* in_sizes, int n_in,
                              void* d_out, int out_size, void* d_ws, size_t ws_size,
                              hipStream_t stream) {
  (void)in_sizes; (void)n_in; (void)out_size; (void)ws_size;

  const float* x     = (const float*)d_in[0];
  const float* mu_w  = (const float*)d_in[1];
  const float* mu_r  = (const float*)d_in[2];
  const float* mu_k  = (const float*)d_in[3];
  const float* mu_v  = (const float*)d_in[4];
  const float* mu_g  = (const float*)d_in[5];
  const float* w0    = (const float*)d_in[6];
  const float* Aw    = (const float*)d_in[7];
  const float* Bw    = (const float*)d_in[8];
  const float* Wr    = (const float*)d_in[9];
  const float* Wk    = (const float*)d_in[10];
  const float* Wv    = (const float*)d_in[11];
  const float* Wg    = (const float*)d_in[12];
  const float* Wo    = (const float*)d_in[13];
  const float* u     = (const float*)d_in[14];
  const float* gng   = (const float*)d_in[15];
  const float* gnb   = (const float*)d_in[16];
  float* out = (float*)d_out;

  char* ws = (char*)d_ws;
  size_t off = 0;
  auto alloc = [&](size_t bytes) { size_t o = off; off += (bytes + 255) & ~(size_t)255; return o; };

  const size_t SZ_H = (size_t)MROWS * DD * sizeof(_Float16);   // 8 MB
  const size_t SZ_F = (size_t)MROWS * DD * sizeof(float);      // 16 MB
  const size_t SZ_W = (size_t)DD * DD * sizeof(_Float16);      // 2 MB
  const size_t SZ_L = (size_t)DD * RLORA * sizeof(_Float16);   // 128 KB
  const size_t SZ_S = (size_t)BHQ * NC * HDIM * HDIM * sizeof(float);  // 16 MB

  _Float16* xw16  = (_Float16*)(ws + alloc(SZ_H));
  _Float16* xr16  = (_Float16*)(ws + alloc(SZ_H));
  _Float16* xk16  = (_Float16*)(ws + alloc(SZ_H));
  _Float16* xv16  = (_Float16*)(ws + alloc(SZ_H));
  _Float16* xg16  = (_Float16*)(ws + alloc(SZ_H));
  _Float16* wAwT  = (_Float16*)(ws + alloc(SZ_L));
  _Float16* wBwT  = (_Float16*)(ws + alloc(SZ_L));
  _Float16* wWrT  = (_Float16*)(ws + alloc(SZ_W));
  _Float16* wWkT  = (_Float16*)(ws + alloc(SZ_W));
  _Float16* wWvT  = (_Float16*)(ws + alloc(SZ_W));
  _Float16* wWgT  = (_Float16*)(ws + alloc(SZ_W));
  _Float16* wWoT  = (_Float16*)(ws + alloc(SZ_W));
  float*    hraw  = (float*)   (ws + alloc((size_t)MROWS * RLORA * sizeof(float)));
  _Float16* h16   = (_Float16*)(ws + alloc((size_t)MROWS * RLORA * sizeof(_Float16)));
  float*    zbuf  = (float*)   (ws + alloc(SZ_F));
  float*    r32   = (float*)   (ws + alloc(SZ_F));
  float*    k32   = (float*)   (ws + alloc(SZ_F));
  float*    v32   = (float*)   (ws + alloc(SZ_F));
  float*    g32   = (float*)   (ws + alloc(SZ_F));
  _Float16* yg16  = (_Float16*)(ws + alloc(SZ_H));
  float*    Spart = (float*)   (ws + alloc(SZ_S));
  float*    Sinit = (float*)   (ws + alloc(SZ_S));
  float*    Pprod = (float*)   (ws + alloc((size_t)BHQ * NC * HDIM * sizeof(float)));

  const int EB = 256;
  k_f32_to_f16_T<<<(DD * RLORA + EB - 1) / EB, EB, 0, stream>>>(Aw, wAwT, DD, RLORA);
  k_f32_to_f16_T<<<(RLORA * DD + EB - 1) / EB, EB, 0, stream>>>(Bw, wBwT, RLORA, DD);
  k_f32_to_f16_T<<<(DD * DD + EB - 1) / EB, EB, 0, stream>>>(Wr, wWrT, DD, DD);
  k_f32_to_f16_T<<<(DD * DD + EB - 1) / EB, EB, 0, stream>>>(Wk, wWkT, DD, DD);
  k_f32_to_f16_T<<<(DD * DD + EB - 1) / EB, EB, 0, stream>>>(Wv, wWvT, DD, DD);
  k_f32_to_f16_T<<<(DD * DD + EB - 1) / EB, EB, 0, stream>>>(Wg, wWgT, DD, DD);
  k_f32_to_f16_T<<<(DD * DD + EB - 1) / EB, EB, 0, stream>>>(Wo, wWoT, DD, DD);

  k_shift_mix<<<(MROWS * DD) / EB, EB, 0, stream>>>(x, mu_w, mu_r, mu_k, mu_v, mu_g,
                                                    xw16, xr16, xk16, xv16, xg16);

  // LoRA decay path: h = tanh(xw @ Aw) ; z = h @ Bw (decay finished inside the scan)
  k_gemm_wmma<4><<<dim3(1, MROWS / 128), 256, 0, stream>>>(xw16, wAwT, hraw, MROWS, RLORA, DD);
  k_tanh_h<<<(MROWS * RLORA) / EB, EB, 0, stream>>>(hraw, h16);
  k_gemm_wmma<8><<<dim3(DD / 128, MROWS / 128), 256, 0, stream>>>(h16, wBwT, zbuf, MROWS, DD, RLORA);

  // projections r,k,v,g(raw)
  k_gemm_wmma<8><<<dim3(DD / 128, MROWS / 128), 256, 0, stream>>>(xr16, wWrT, r32, MROWS, DD, DD);
  k_gemm_wmma<8><<<dim3(DD / 128, MROWS / 128), 256, 0, stream>>>(xk16, wWkT, k32, MROWS, DD, DD);
  k_gemm_wmma<8><<<dim3(DD / 128, MROWS / 128), 256, 0, stream>>>(xv16, wWvT, v32, MROWS, DD, DD);
  k_gemm_wmma<8><<<dim3(DD / 128, MROWS / 128), 256, 0, stream>>>(xg16, wWgT, g32, MROWS, DD, DD);

  // chunked WKV: partials -> prefix -> replay (+ decay + GroupNorm + SiLU-gate) -> f16
  k_scan_partial<<<dim3(BHQ, NC), HDIM, 0, stream>>>(k32, v32, zbuf, w0, Spart, Pprod);
  k_scan_prefix<<<BHQ, HDIM, 0, stream>>>(Spart, Pprod, Sinit);
  k_scan_final<<<dim3(BHQ, NC), HDIM, 0, stream>>>(r32, k32, v32, zbuf, g32, w0, u, gng, gnb,
                                                   Sinit, yg16);

  // output projection
  k_gemm_wmma<8><<<dim3(DD / 128, MROWS / 128), 256, 0, stream>>>(yg16, wWoT, out, MROWS, DD, DD);
}
